// UncertaintyDCN_6871947673756
// MI455X (gfx1250) — matrix operand order repaired and checked by reading, compiled, not run on hardware
//
#include <hip/hip_runtime.h>

// ---------------- problem constants (from reference) ----------------
#define CC    256      // channels
#define GG    4        // groups
#define GCC   64       // channels per group
#define KK    3
#define PP    9        // K*K points
#define HH    96
#define WW    96
#define HP    98       // H + 2*PAD
#define WP    98
#define NCOL  108      // G*P*3
#define NPAD  112      // padded to multiple of 16 for WMMA

typedef _Float16 v16h __attribute__((ext_vector_type(16)));
typedef float    v8f  __attribute__((ext_vector_type(8)));

// ---------------------------------------------------------------
// 1) pad + NCHW->NHWC transpose of x into x_pad [N, HP, WP, C]
// ---------------------------------------------------------------
__global__ void pad_nhwc_kernel(const float* __restrict__ x, float* __restrict__ xpad, int total) {
    int idx = blockIdx.x * blockDim.x + threadIdx.x;
    if (idx >= total) return;
    int c  = idx & (CC - 1);
    int t  = idx >> 8;            // /256
    int xx = t % WP;
    t /= WP;
    int yy = t % HP;
    int n  = t / HP;
    float v = 0.0f;
    if (yy >= 1 && yy <= HH && xx >= 1 && xx <= WW) {
        v = x[(((size_t)n * CC + c) * HH + (yy - 1)) * WW + (xx - 1)];
    }
    xpad[idx] = v;
}

// ---------------------------------------------------------------
// 2) depthwise 3x3 conv + bias -> ctx (NHWC), block-reduced sum/sumsq atomics
//    one block = one (n, c, h) row, 128 threads (96 active over w)
// ---------------------------------------------------------------
__global__ void dwconv_kernel(const float* __restrict__ x, const float* __restrict__ dww,
                              const float* __restrict__ dwb, float* __restrict__ ctx,
                              float* __restrict__ sums) {
    int h = blockIdx.x % HH;
    int c = (blockIdx.x / HH) % CC;
    int n = blockIdx.x / (HH * CC);
    int w = threadIdx.x;
    float val = 0.0f;
    if (w < WW) {
        val = dwb[c];
        const float* xc = x + ((size_t)n * CC + c) * HH * WW;
        #pragma unroll
        for (int ky = 0; ky < KK; ++ky) {
            int yy = h + ky - 1;
            if (yy < 0 || yy >= HH) continue;
            #pragma unroll
            for (int kx = 0; kx < KK; ++kx) {
                int xx = w + kx - 1;
                if (xx < 0 || xx >= WW) continue;
                val += xc[yy * WW + xx] * dww[c * 9 + ky * 3 + kx];
            }
        }
        ctx[(((size_t)n * HH + h) * WW + w) * CC + c] = val;
    }
    __shared__ float s1[128];
    __shared__ float s2[128];
    int tid = threadIdx.x;
    s1[tid] = (w < WW) ? val : 0.0f;
    s2[tid] = (w < WW) ? val * val : 0.0f;
    __syncthreads();
    #pragma unroll
    for (int s = 64; s > 0; s >>= 1) {
        if (tid < s) { s1[tid] += s1[tid + s]; s2[tid] += s2[tid + s]; }
        __syncthreads();
    }
    if (tid == 0) {
        atomicAdd(&sums[2 * n + 0], s1[0]);
        atomicAdd(&sums[2 * n + 1], s2[0]);
    }
}

// ---------------------------------------------------------------
// 3) finalize GroupNorm(1,C) stats per sample
// ---------------------------------------------------------------
__global__ void gn_finalize_kernel(const float* __restrict__ sums, float* __restrict__ params, int N) {
    int n = threadIdx.x;
    if (n < N) {
        const float cnt = (float)CC * HH * WW;
        float mu  = sums[2 * n + 0] / cnt;
        float var = sums[2 * n + 1] / cnt - mu * mu;
        params[2 * n + 0] = mu;
        params[2 * n + 1] = rsqrtf(var + 1e-5f);
    }
}

// ---------------------------------------------------------------
// 4) normalize + affine + exact GELU -> feat f16 (NHWC)
// ---------------------------------------------------------------
__global__ void norm_gelu_kernel(const float* __restrict__ ctx, const float* __restrict__ params,
                                 const float* __restrict__ gnw, const float* __restrict__ gnb,
                                 _Float16* __restrict__ feat, int total) {
    int idx = blockIdx.x * blockDim.x + threadIdx.x;
    if (idx >= total) return;
    int c = idx & (CC - 1);
    int n = idx / (HH * WW * CC);
    float mu = params[2 * n + 0];
    float iv = params[2 * n + 1];
    float v = (ctx[idx] - mu) * iv * gnw[c] + gnb[c];
    float ge = 0.5f * v * (1.0f + erff(v * 0.70710678118654752f));
    feat[idx] = (_Float16)ge;
}

// ---------------------------------------------------------------
// 5) om_w fp32 [108,256] -> f16, zero-padded to [112,256]
// ---------------------------------------------------------------
__global__ void convert_w_kernel(const float* __restrict__ w, _Float16* __restrict__ wf16) {
    int idx = blockIdx.x * blockDim.x + threadIdx.x;
    if (idx >= NPAD * CC) return;
    int o = idx >> 8;
    wf16[idx] = (o < NCOL) ? (_Float16)w[idx] : (_Float16)0.0f;
}

// ---------------------------------------------------------------
// 6) WMMA GEMM: om[M,112] = feat[M,256] @ om_w^T + bias
//    one wave computes a 16(M) x 112(N) strip; K=256 in 8 steps of 32
// ---------------------------------------------------------------
__global__ void gemm_wmma_kernel(const _Float16* __restrict__ A, const _Float16* __restrict__ Bw,
                                 const float* __restrict__ bias, float* __restrict__ Om, int Mtiles) {
    int wave = threadIdx.x >> 5;
    int lane = threadIdx.x & 31;
    int tile = blockIdx.x * 8 + wave;
    if (tile >= Mtiles) return;
    int m0  = tile << 4;
    int hi  = lane >> 4;        // 0 | 1 half-wave
    int lm  = lane & 15;
    const _Float16* arow = A + (size_t)(m0 + lm) * CC;
    int aoff = hi * 8;          // A frag: K split 0..7 / 8..15 across half-waves
    int boff = hi * 8;          // B frag: K split 0..7 / 8..15 across half-waves

    v8f acc[7];
    #pragma unroll
    for (int t = 0; t < 7; ++t)
        #pragma unroll
        for (int i = 0; i < 8; ++i) acc[t][i] = 0.0f;

    #pragma unroll
    for (int kb = 0; kb < CC; kb += 32) {
        v16h a;
        const _Float16* ap = arow + kb + aoff;
        #pragma unroll
        for (int e = 0; e < 8; ++e) { a[e] = ap[e]; a[8 + e] = ap[16 + e]; }
        #pragma unroll
        for (int t = 0; t < 7; ++t) {
            const _Float16* bp = Bw + (size_t)(t * 16 + lm) * CC + kb + boff;
            v16h b;
            #pragma unroll
            for (int e = 0; e < 8; ++e) { b[e] = bp[e]; b[8 + e] = bp[16 + e]; }
            acc[t] = __builtin_amdgcn_wmma_f32_16x16x32_f16(
                false, a, false, b, (short)0, acc[t], false, false);
        }
    }
    #pragma unroll
    for (int t = 0; t < 7; ++t) {
        int col = t * 16 + lm;
        float bv = (col < NCOL) ? bias[col] : 0.0f;
        #pragma unroll
        for (int r = 0; r < 8; ++r) {
            int row = m0 + hi * 8 + r;
            Om[(size_t)row * NPAD + col] = acc[t][r] + bv;
        }
    }
}

// ---------------------------------------------------------------
// 7) softmax(mask)*u + DCNv3 bilinear sampling -> out NCHW
//    one wave per (n,h,w,g); lane owns channels (lane, lane+32) of the group
// ---------------------------------------------------------------
__global__ void dcn_sample_kernel(const float* __restrict__ xpad, const float* __restrict__ Om,
                                  const float* __restrict__ um, float* __restrict__ out) {
    int gid  = blockIdx.x * 8 + (threadIdx.x >> 5);
    int lane = threadIdx.x & 31;
    int g = gid & 3;
    int t = gid >> 2;
    int w = t % WW;
    t /= WW;
    int h = t % HH;
    int n = t / HH;

    const float* omr = Om + (((size_t)n * HH + h) * WW + w) * NPAD;
    float u = um[((size_t)n * HH + h) * WW + w];

    float ml[PP];
    float mx = -3.0e38f;
    #pragma unroll
    for (int p = 0; p < PP; ++p) {
        ml[p] = omr[2 * GG * PP + g * PP + p];
        mx = fmaxf(mx, ml[p]);
    }
    float s = 0.0f;
    #pragma unroll
    for (int p = 0; p < PP; ++p) { ml[p] = expf(ml[p] - mx); s += ml[p]; }
    float scale = u / s;

    const float* xb = xpad + (size_t)n * HP * WP * CC;
    int c0 = g * GCC + lane;
    float a0 = 0.0f, a1 = 0.0f;

    #pragma unroll
    for (int p = 0; p < PP; ++p) {
        float dx = omr[g * (PP * 2) + 2 * p + 0];
        float dy = omr[g * (PP * 2) + 2 * p + 1];
        // padded-image pixel coords (align_corners=False math collapses to this)
        float sx = (float)(w + p / 3) + dx;
        float sy = (float)(h + p % 3) + dy;
        float x0f = floorf(sx), y0f = floorf(sy);
        float wx = sx - x0f,    wy = sy - y0f;
        int x0 = (int)x0f, y0 = (int)y0f;
        float wp = ml[p] * scale;
        #pragma unroll
        for (int dyi = 0; dyi < 2; ++dyi) {
            #pragma unroll
            for (int dxi = 0; dxi < 2; ++dxi) {
                int yy = y0 + dyi, xx = x0 + dxi;
                float cw = (dyi ? wy : 1.0f - wy) * (dxi ? wx : 1.0f - wx) * wp;
                float vf = (yy >= 0 && yy < HP && xx >= 0 && xx < WP) ? 1.0f : 0.0f;
                int yc = min(max(yy, 0), HP - 1);
                int xc = min(max(xx, 0), WP - 1);
                const float* src = xb + ((size_t)yc * WP + xc) * CC + c0;
                a0 += cw * vf * src[0];
                a1 += cw * vf * src[32];
            }
        }
    }
    out[(((size_t)n * CC + c0) * HH + h) * WW + w]        = a0;
    out[(((size_t)n * CC + c0 + 32) * HH + h) * WW + w]   = a1;
}

// ---------------------------------------------------------------
extern "C" void kernel_launch(void* const* d_in, const int* in_sizes, int n_in,
                              void* d_out, int out_size, void* d_ws, size_t ws_size,
                              hipStream_t stream) {
    const float* x    = (const float*)d_in[0];  // [N,256,96,96]
    const float* um   = (const float*)d_in[1];  // [N,1,96,96]
    const float* dw_w = (const float*)d_in[2];  // [256,1,3,3]
    const float* dw_b = (const float*)d_in[3];  // [256]
    const float* gn_w = (const float*)d_in[4];  // [256]
    const float* gn_b = (const float*)d_in[5];  // [256]
    const float* om_w = (const float*)d_in[6];  // [108,256]
    const float* om_b = (const float*)d_in[7];  // [108]
    float* out = (float*)d_out;

    const int N = in_sizes[0] / (CC * HH * WW);
    const size_t NHW = (size_t)N * HH * WW;

    // ---- workspace carve-up (all region sizes multiples of 256B) ----
    char* ws = (char*)d_ws;
    size_t off = 0;
    float*    xpad   = (float*)(ws + off);  off += (size_t)N * HP * WP * CC * 4;
    float*    ctx    = (float*)(ws + off);  off += NHW * CC * 4;
    _Float16* feat   = (_Float16*)(ws + off); off += NHW * CC * 2;
    float*    om     = (float*)(ws + off);  off += NHW * NPAD * 4;
    _Float16* wf16   = (_Float16*)(ws + off); off += (size_t)NPAD * CC * 2;
    float*    sums   = (float*)(ws + off);  off += 256;
    float*    params = (float*)(ws + off);  off += 256;
    (void)ws_size; (void)n_in; (void)out_size;

    hipMemsetAsync(sums, 0, 2 * N * sizeof(float), stream);

    // 1) pad + transpose x -> NHWC padded
    {
        int total = N * HP * WP * CC;
        pad_nhwc_kernel<<<(total + 255) / 256, 256, 0, stream>>>(x, xpad, total);
    }
    // 2) depthwise conv + stats
    dwconv_kernel<<<N * CC * HH, 128, 0, stream>>>(x, dw_w, dw_b, ctx, sums);
    // 3) GN stats
    gn_finalize_kernel<<<1, 32, 0, stream>>>(sums, params, N);
    // 4) normalize + GELU -> f16
    {
        int total = (int)(NHW * CC);
        norm_gelu_kernel<<<(total + 255) / 256, 256, 0, stream>>>(ctx, params, gn_w, gn_b, feat, total);
    }
    // 5) weights -> f16 (padded)
    convert_w_kernel<<<(NPAD * CC + 255) / 256, 256, 0, stream>>>(om_w, wf16);
    // 6) WMMA GEMM: [NHW,256] x [256,108] + bias
    {
        int Mtiles = (int)(NHW / 16);   // NHW = N*9216, divisible by 16
        gemm_wmma_kernel<<<(Mtiles + 7) / 8, 256, 0, stream>>>(feat, wf16, om_b, om, Mtiles);
    }
    // 7) sampling: one wave per (n,h,w,g)
    {
        int waves = (int)(NHW * GG);    // divisible by 8
        dcn_sample_kernel<<<waves / 8, 256, 0, stream>>>(xpad, om, um, out);
    }
}
